// SparseLinear_38096359916173
// MI455X (gfx1250) — matrix operand order repaired and checked by reading, not compiled, not run on hardware
//
#include <hip/hip_runtime.h>

typedef __attribute__((ext_vector_type(16))) __bf16 bf16x16;
typedef __attribute__((ext_vector_type(8)))  __bf16 bf16x8;
typedef __attribute__((ext_vector_type(4)))  __bf16 bf16x4;
typedef __attribute__((ext_vector_type(8)))  float  f32x8;
typedef __attribute__((ext_vector_type(4)))  float  f32x4v;

#define TM 128
#define TN 128
#define TK 32
#define SAS 48   // halfword stride: 96B per row -> b128 runs stay 16B-aligned
#define SBS 48

// A fragment: two 8-element K-runs per lane (K = 8h..8h+7 and 16+8h..16+8h+7).
// B fragment: two 8-element K-runs per lane (K = 8h..8h+7 and 16+8h..16+8h+7).
union Frag { bf16x16 v; bf16x8 h[2]; };

// 256 threads = 8 wave32; wave grid 2(M) x 4(N); each wave -> 64x32 output
// (4x2 subtiles of 16x16). fp32 via bf16 hi/lo split done ONCE at LDS-store
// time; fragments are pure ds_load_b128, zero per-element shuffling.
__global__ __launch_bounds__(256)
void sparse_linear_wmma(const float* __restrict__ X,
                        const float* __restrict__ Mk,
                        const float* __restrict__ W,
                        const float* __restrict__ Bias,
                        float* __restrict__ Out,
                        int Din, int Dout)
{
    __shared__ __bf16 sAh[TM * SAS];   // x hi:          [row m][k]
    __shared__ __bf16 sAl[TM * SAS];   // x lo
    __shared__ __bf16 sBh[TN * SBS];   // (mask*W)^T hi: [col n][k]
    __shared__ __bf16 sBl[TN * SBS];   // (mask*W)^T lo

    const int tid  = threadIdx.x;
    const int lane = tid & 31;
    const int wave = tid >> 5;
    const int hiH  = lane >> 4;        // K-half selector per fragment layout
    const int l16  = lane & 15;

    const int blockM = blockIdx.y * TM;
    const int blockN = blockIdx.x * TN;
    const int waveM  = (wave >> 2) * 64;
    const int waveN  = (wave & 3) * 32;

    // Per-thread global-load assignments: 1024 float4 per tile / 256 threads.
    int arow[4], acol[4], bkr[4], bn0[4];
#pragma unroll
    for (int i = 0; i < 4; ++i) {
        int idx = tid + i * 256;
        arow[i] = idx >> 3;            // A: 8 float4 per 32-wide row
        acol[i] = (idx & 7) * 4;
        bkr[i]  = idx >> 5;            // B: 32 float4 per 128-wide k-row
        bn0[i]  = (idx & 31) * 4;
    }

    f32x4v rA[4], rW[4], rM[4];        // register double-buffer for global tiles

    auto loadTiles = [&](int k0) {
#pragma unroll
        for (int i = 0; i < 4; ++i) {
            rA[i] = *(const f32x4v*)(X + (size_t)(blockM + arow[i]) * Din + k0 + acol[i]);
            size_t bb = (size_t)(k0 + bkr[i]) * Dout + blockN + bn0[i];
            rW[i] = *(const f32x4v*)(W  + bb);
            rM[i] = *(const f32x4v*)(Mk + bb);
        }
    };

    // Split to bf16 hi/lo exactly once per element, at staging time.
    auto storeTiles = [&]() {
#pragma unroll
        for (int i = 0; i < 4; ++i) {
            bf16x4 ah, al;
#pragma unroll
            for (int j = 0; j < 4; ++j) {
                float v = rA[i][j];
                __bf16 h = (__bf16)v;
                ah[j] = h;
                al[j] = (__bf16)(v - (float)h);
            }
            int aoff = arow[i] * SAS + acol[i];
            *(bf16x4*)(&sAh[aoff]) = ah;   // ds_store_b64
            *(bf16x4*)(&sAl[aoff]) = al;

            f32x4v p = rW[i] * rM[i];      // fuse the 0/1 mask (exact)
#pragma unroll
            for (int j = 0; j < 4; ++j) {  // transpose into [n][k] (b16 scatter)
                float v = p[j];
                __bf16 h = (__bf16)v;
                int boff = (bn0[i] + j) * SBS + bkr[i];
                sBh[boff] = h;
                sBl[boff] = (__bf16)(v - (float)h);
            }
        }
    };

    f32x8 acc[4][2] = {};

    loadTiles(0);

    for (int k0 = 0; k0 < Din; k0 += TK) {
        storeTiles();
        __syncthreads();
        if (k0 + TK < Din) loadTiles(k0 + TK);   // overlap next tile with WMMAs

        // ---- B fragments: two split K-runs per lane (K = 8h.., 16+8h..) ----
        Frag bh[2], bl[2];
#pragma unroll
        for (int nt = 0; nt < 2; ++nt) {
            int off = (waveN + nt * 16 + l16) * SBS + 8 * hiH;
            bh[nt].h[0] = *(const bf16x8*)(&sBh[off]);
            bh[nt].h[1] = *(const bf16x8*)(&sBh[off + 16]);
            bl[nt].h[0] = *(const bf16x8*)(&sBl[off]);
            bl[nt].h[1] = *(const bf16x8*)(&sBl[off + 16]);
        }

        // ---- A fragments: two split K-runs per lane (K = 8h.., 16+8h..) ----
#pragma unroll
        for (int mt = 0; mt < 4; ++mt) {
            int off = (waveM + mt * 16 + l16) * SAS + 8 * hiH;
            Frag ah, al;
            ah.h[0] = *(const bf16x8*)(&sAh[off]);
            ah.h[1] = *(const bf16x8*)(&sAh[off + 16]);
            al.h[0] = *(const bf16x8*)(&sAl[off]);
            al.h[1] = *(const bf16x8*)(&sAl[off + 16]);
#pragma unroll
            for (int nt = 0; nt < 2; ++nt) {
                acc[mt][nt] = __builtin_amdgcn_wmma_f32_16x16x32_bf16(
                    false, ah.v, false, bh[nt].v, (short)0, acc[mt][nt], false, false);
                acc[mt][nt] = __builtin_amdgcn_wmma_f32_16x16x32_bf16(
                    false, ah.v, false, bl[nt].v, (short)0, acc[mt][nt], false, false);
                acc[mt][nt] = __builtin_amdgcn_wmma_f32_16x16x32_bf16(
                    false, al.v, false, bh[nt].v, (short)0, acc[mt][nt], false, false);
            }
        }
        __syncthreads();
    }

    // ---- epilogue: fused bias add, coalesced stores ----
#pragma unroll
    for (int nt = 0; nt < 2; ++nt) {
        const int gn = blockN + waveN + nt * 16 + l16;
        const float bias = Bias[gn];
#pragma unroll
        for (int mt = 0; mt < 4; ++mt) {
            const int gm0 = blockM + waveM + mt * 16 + 8 * hiH; // C/D: m = r + 8*(lane>=16)
#pragma unroll
            for (int r = 0; r < 8; ++r)
                Out[(size_t)(gm0 + r) * Dout + gn] = acc[mt][nt][r] + bias;
        }
    }
}

extern "C" void kernel_launch(void* const* d_in, const int* in_sizes, int n_in,
                              void* d_out, int out_size, void* d_ws, size_t ws_size,
                              hipStream_t stream) {
    const float* x    = (const float*)d_in[0];
    const float* mask = (const float*)d_in[1];
    const float* W    = (const float*)d_in[2];
    const float* bias = (const float*)d_in[3];
    float* out = (float*)d_out;

    const int Dout = in_sizes[3];                 // 2048
    const int Din  = in_sizes[1] / Dout;          // 4096
    const int B    = in_sizes[0] / Din;           // 4096

    dim3 grid(Dout / TN, B / TM);                 // (16, 32)
    sparse_linear_wmma<<<grid, 256, 0, stream>>>(x, mask, W, bias, out, Din, Dout);
}
